// SubjectAwareAttention_52974126629160
// MI455X (gfx1250) — matrix operand order repaired and checked by reading, not compiled, not run on hardware
//
#include <hip/hip_runtime.h>
#include <hip/hip_bf16.h>

typedef __bf16 bf16_t;
typedef __attribute__((ext_vector_type(16))) __bf16 v16bf;
typedef __attribute__((ext_vector_type(8)))  __bf16 bf16x8;
typedef __attribute__((ext_vector_type(8)))  float  v8f;

static constexpr int Bb  = 2;
static constexpr int Ll  = 2048;
static constexpr int Dd  = 1024;
static constexpr int Hh  = 16;
static constexpr int HDd = 64;
static constexpr int DSs = 64;

#define DEV static __device__ __forceinline__

DEV bf16_t f2bf(float f) {
  union { float f; unsigned u; } uf; uf.f = f;
  unsigned r = uf.u + 0x7FFFu + ((uf.u >> 16) & 1u);
  unsigned short h = (unsigned short)(r >> 16);
  bf16_t b; __builtin_memcpy(&b, &h, 2); return b;
}

DEV v16bf frag_from2(bf16x8 lo, bf16x8 hi) {
  v16bf v;
#pragma unroll
  for (int i = 0; i < 8; ++i) { v[i] = lo[i]; v[i + 8] = hi[i]; }
  return v;
}

// CDNA5 async global->LDS copy (16B per lane). Tracked by ASYNCcnt.
// Generic LDS pointer's low 32 bits are the LDS byte offset (ISA 10.2 aperture
// truncation), so passing them as the "vdst" LDS-address VGPR is valid.
DEV void async_load_b128(const void* gptr, void* lptr) {
  unsigned lds = (unsigned)(unsigned long long)(uintptr_t)lptr;
  unsigned long long g = (unsigned long long)(uintptr_t)gptr;
  asm volatile("global_load_async_to_lds_b128 %0, %1, off"
               :: "v"(lds), "v"(g) : "memory");
}
DEV void wait_async0() { asm volatile("s_wait_asynccnt 0x0" ::: "memory"); }

// ---------------------------------------------------------------------------
// fp32 -> bf16 elementwise convert
// ---------------------------------------------------------------------------
__global__ void cvt_bf16_kernel(const float* __restrict__ src,
                                bf16_t* __restrict__ dst, int n) {
  int i = blockIdx.x * blockDim.x + threadIdx.x;
  if (i < n) dst[i] = f2bf(src[i]);
}

// ---------------------------------------------------------------------------
// subject bias: qb[b][n] = subj[b,:] @ Wsq[:,n] + bsq[n]  (and kb likewise)
// ---------------------------------------------------------------------------
__global__ void subject_bias_kernel(const float* __restrict__ subj,
                                    const float* __restrict__ Wsq,
                                    const float* __restrict__ bsq,
                                    const float* __restrict__ Wsk,
                                    const float* __restrict__ bsk,
                                    float* __restrict__ qb,
                                    float* __restrict__ kb) {
  int i = blockIdx.x * blockDim.x + threadIdx.x;
  if (i >= Bb * Dd) return;
  int b = i / Dd, n = i % Dd;
  float aq = bsq[n], ak = bsk[n];
#pragma unroll 8
  for (int s = 0; s < DSs; ++s) {
    float e = subj[b * DSs + s];
    aq += e * Wsq[s * Dd + n];
    ak += e * Wsk[s * Dd + n];
  }
  qb[i] = aq; kb[i] = ak;
}

// ---------------------------------------------------------------------------
// bf16 WMMA GEMM: C[M,N] = A[M,K] @ W[K,N] (+bias)
//   mode 0: bias = per-batch row bias [Bb][N] (may be null); out bf16 in
//           [B,H,L,HD] "head" layout for the attention stage.
//   mode 1: bias = per-column bias [N]; out fp32 row-major [M,N].
// Block: 128 threads (4 waves). Tile 64x64, K-step 32.
// A tile staged via global_load_async_to_lds_b128; W tile transposed via VGPRs.
// ---------------------------------------------------------------------------
__global__ __launch_bounds__(128)
void gemm_bf16_kernel(const bf16_t* __restrict__ A,
                      const bf16_t* __restrict__ W,
                      const float* __restrict__ bias,
                      void* __restrict__ out,
                      int M, int N, int K, int mode) {
  __shared__ __align__(16) bf16_t Asub[64 * 32];   // [m][k]
  __shared__ __align__(16) bf16_t Bt[64 * 32];     // [n][k] (transposed W tile)

  const int nTilesN = N / 64;
  const int m0 = (blockIdx.x / nTilesN) * 64;
  const int n0 = (blockIdx.x % nTilesN) * 64;

  const int wv    = threadIdx.x >> 5;
  const int lane  = threadIdx.x & 31;
  const int laneN = lane & 15;
  const int hf    = lane >> 4;

  const int aRow = threadIdx.x >> 1;            // 0..63
  const int aKp  = (threadIdx.x & 1) * 16;      // 0 or 16
  const int wKl  = threadIdx.x & 31;            // 0..31
  const int wNp  = (threadIdx.x >> 5) * 16;     // 0,16,32,48

  v8f acc[4];
  const v8f zero = {0.f, 0.f, 0.f, 0.f, 0.f, 0.f, 0.f, 0.f};
#pragma unroll
  for (int j = 0; j < 4; ++j) acc[j] = zero;

  for (int kk = 0; kk < K; kk += 32) {
    __syncthreads();
    {   // A tile 64x32: async global -> LDS, 32B per thread
      const bf16_t* src = A + (size_t)(m0 + aRow) * K + kk + aKp;
      async_load_b128(src,     &Asub[aRow * 32 + aKp]);
      async_load_b128(src + 8, &Asub[aRow * 32 + aKp + 8]);
    }
    {   // W tile: load row-major, scatter transposed into Bt[n][k]
      const bf16_t* src = W + (size_t)(kk + wKl) * N + n0 + wNp;
      bf16x8 v0 = *(const bf16x8*)(src);
      bf16x8 v1 = *(const bf16x8*)(src + 8);
#pragma unroll
      for (int e = 0; e < 8; ++e) {
        Bt[(wNp + e) * 32 + wKl]     = v0[e];
        Bt[(wNp + 8 + e) * 32 + wKl] = v1[e];
      }
    }
    if (kk + 32 < K) {  // prefetch next tiles into cache hierarchy
      __builtin_prefetch(A + (size_t)(m0 + aRow) * K + kk + 32 + aKp, 0, 1);
      __builtin_prefetch(W + (size_t)(kk + 32 + wKl) * N + n0 + wNp, 0, 1);
    }
    wait_async0();
    __syncthreads();

    const int mloc = wv * 16 + laneN;
    v16bf av = frag_from2(*(const bf16x8*)(&Asub[mloc * 32 + 8 * hf]),
                          *(const bf16x8*)(&Asub[mloc * 32 + 16 + 8 * hf]));
    v16bf bv[4];
#pragma unroll
    for (int j = 0; j < 4; ++j) {
      int nn = j * 16 + laneN;
      bv[j] = frag_from2(*(const bf16x8*)(&Bt[nn * 32 + 8 * hf]),
                         *(const bf16x8*)(&Bt[nn * 32 + 16 + 8 * hf]));
    }
#pragma unroll
    for (int j = 0; j < 4; ++j)
      acc[j] = __builtin_amdgcn_wmma_f32_16x16x32_bf16(
          false, av, false, bv[j], (short)0, acc[j], false, false);
  }

  const int rowBase = m0 + wv * 16 + 8 * hf;   // C layout: m = r + 8*hf
#pragma unroll
  for (int j = 0; j < 4; ++j) {
    int col = n0 + j * 16 + laneN;
#pragma unroll
    for (int r = 0; r < 8; ++r) {
      int row = rowBase + r;
      float v = acc[j][r];
      if (mode == 0) {
        if (bias) v += bias[(row / Ll) * N + col];
        int b = row / Ll, l = row % Ll;
        int h = col >> 6, hd = col & 63;
        ((bf16_t*)out)[(((size_t)(b * Hh + h) * Ll + l) << 6) + hd] = f2bf(v);
      } else {
        v += bias[col];
        ((float*)out)[(size_t)row * N + col] = v;
      }
    }
  }
}

// ---------------------------------------------------------------------------
// Flash attention: per (b,h), 64 query rows per block (16 per wave),
// online softmax over 64-key blocks. Q/K/V bf16 in [B,H,L,64].
// K block staged via async global->LDS (no transpose needed); V transposed.
// Output attnO bf16 in [B,L,D] layout (ready for output projection).
// ---------------------------------------------------------------------------
__global__ __launch_bounds__(128)
void flash_attn_kernel(const bf16_t* __restrict__ Q,
                       const bf16_t* __restrict__ Kk,
                       const bf16_t* __restrict__ V,
                       const unsigned char* __restrict__ mask,
                       bf16_t* __restrict__ O) {
  __shared__ __align__(16) bf16_t Ksub[64 * 64];      // [key][hd]
  __shared__ __align__(16) bf16_t Vt[64 * 64];        // [hd][key]
  __shared__ __align__(16) bf16_t Plds[4][16 * 64];   // per-wave P tile [m][key]

  const int qblk = blockIdx.x & 31;          // L/64 = 32 query blocks
  const int bh   = blockIdx.x >> 5;          // B*H = 32
  const int b = bh >> 4, h = bh & 15;

  const int wv    = threadIdx.x >> 5;
  const int lane  = threadIdx.x & 31;
  const int laneN = lane & 15;
  const int hf    = lane >> 4;

  const size_t headBase = (size_t)(b * Hh + h) * Ll * HDd;
  const int qrow0 = qblk * 64 + wv * 16;
  const float scale = 0.125f;                // 1/sqrt(64)

  // Q fragments: 2 chunks of 16x32 over HD=64
  v16bf qa[2];
  {
    const bf16_t* qp = Q + headBase + (size_t)(qrow0 + laneN) * HDd;
#pragma unroll
    for (int c = 0; c < 2; ++c)
      qa[c] = frag_from2(*(const bf16x8*)(qp + c * 32 + 8 * hf),
                         *(const bf16x8*)(qp + c * 32 + 16 + 8 * hf));
  }

  const v8f zero = {0.f, 0.f, 0.f, 0.f, 0.f, 0.f, 0.f, 0.f};
  v8f o[4];
  float mrow[8], lrow[8];
#pragma unroll
  for (int j = 0; j < 4; ++j) o[j] = zero;
#pragma unroll
  for (int r = 0; r < 8; ++r) { mrow[r] = -3.0e38f; lrow[r] = 0.f; }

  const int sKl = threadIdx.x >> 1;          // staging: key row 0..63
  const int sDp = (threadIdx.x & 1) * 32;    // staging: dim part 0/32

  for (int kb = 0; kb < Ll / 64; ++kb) {
    __syncthreads();   // everyone done with previous Ksub/Vt
    {   // K block: async copy 64x64 bf16 (64B per thread), layout unchanged
      const bf16_t* src = Kk + headBase + (size_t)(kb * 64 + sKl) * HDd + sDp;
#pragma unroll
      for (int q4 = 0; q4 < 4; ++q4)
        async_load_b128(src + q4 * 8, &Ksub[sKl * 64 + sDp + q4 * 8]);
    }
    {   // V block: stage transposed into Vt[hd][key]
      const bf16_t* src = V + headBase + (size_t)(kb * 64 + sKl) * HDd + sDp;
#pragma unroll
      for (int q4 = 0; q4 < 4; ++q4) {
        bf16x8 v = *(const bf16x8*)(src + q4 * 8);
#pragma unroll
        for (int e = 0; e < 8; ++e) Vt[(sDp + q4 * 8 + e) * 64 + sKl] = v[e];
      }
    }
    wait_async0();
    __syncthreads();   // Ksub + Vt visible to all waves

    // S = Q @ K^T from LDS-staged K block
    v8f s[4];
#pragma unroll
    for (int j = 0; j < 4; ++j) {
      const bf16_t* kp = &Ksub[(j * 16 + laneN) * 64 + 8 * hf];
      v16bf kv0 = frag_from2(*(const bf16x8*)(kp),      *(const bf16x8*)(kp + 16));
      v16bf kv1 = frag_from2(*(const bf16x8*)(kp + 32), *(const bf16x8*)(kp + 48));
      s[j] = __builtin_amdgcn_wmma_f32_16x16x32_bf16(
          false, qa[0], false, kv0, (short)0, zero, false, false);
      s[j] = __builtin_amdgcn_wmma_f32_16x16x32_bf16(
          false, qa[1], false, kv1, (short)0, s[j], false, false);
    }

    float mbias[4];
#pragma unroll
    for (int j = 0; j < 4; ++j) {
      int key = kb * 64 + j * 16 + laneN;
      mbias[j] = mask[b * Ll + key] ? -3.0e38f : 0.f;
    }

    // online softmax per row (row r lives across 16 lanes of a half-wave)
    float alpha[8];
#pragma unroll
    for (int r = 0; r < 8; ++r) {
      float rm = -3.0e38f;
#pragma unroll
      for (int j = 0; j < 4; ++j) {
        float sv = s[j][r] * scale + mbias[j];
        s[j][r] = sv;
        rm = fmaxf(rm, sv);
      }
#pragma unroll
      for (int off = 1; off < 16; off <<= 1)
        rm = fmaxf(rm, __shfl_xor(rm, off, 16));
      float mn = fmaxf(mrow[r], rm);
      float a  = __expf(mrow[r] - mn);
      float rs = 0.f;
#pragma unroll
      for (int j = 0; j < 4; ++j) {
        float p = __expf(s[j][r] - mn);
        s[j][r] = p;
        rs += p;
      }
#pragma unroll
      for (int off = 1; off < 16; off <<= 1)
        rs += __shfl_xor(rs, off, 16);
      lrow[r] = lrow[r] * a + rs;
      mrow[r] = mn;
      alpha[r] = a;
    }

    // P (C-layout fp32) -> per-wave LDS as bf16 [m][key]; wave-local, LDS
    // ops from the same wave stay in order (no barrier needed before reuse)
#pragma unroll
    for (int j = 0; j < 4; ++j)
#pragma unroll
      for (int r = 0; r < 8; ++r)
        Plds[wv][(r + 8 * hf) * 64 + j * 16 + laneN] = f2bf(s[j][r]);

    // P A-fragments (2 chunks over the 64 keys)
    v16bf pa[2];
#pragma unroll
    for (int c = 0; c < 2; ++c)
      pa[c] = frag_from2(
          *(const bf16x8*)(&Plds[wv][laneN * 64 + c * 32 + 8 * hf]),
          *(const bf16x8*)(&Plds[wv][laneN * 64 + c * 32 + 16 + 8 * hf]));

    // O = O*alpha + P @ V
#pragma unroll
    for (int j = 0; j < 4; ++j) {
#pragma unroll
      for (int r = 0; r < 8; ++r) o[j][r] *= alpha[r];
      const bf16_t* vp = &Vt[(j * 16 + laneN) * 64 + 8 * hf];
      v16bf vb0 = frag_from2(*(const bf16x8*)(vp),      *(const bf16x8*)(vp + 16));
      v16bf vb1 = frag_from2(*(const bf16x8*)(vp + 32), *(const bf16x8*)(vp + 48));
      o[j] = __builtin_amdgcn_wmma_f32_16x16x32_bf16(
          false, pa[0], false, vb0, (short)0, o[j], false, false);
      o[j] = __builtin_amdgcn_wmma_f32_16x16x32_bf16(
          false, pa[1], false, vb1, (short)0, o[j], false, false);
    }
  }

  // normalize + write attnO [B, L, D]
#pragma unroll
  for (int j = 0; j < 4; ++j)
#pragma unroll
    for (int r = 0; r < 8; ++r) {
      int row = qrow0 + r + 8 * hf;
      int col = h * 64 + j * 16 + laneN;
      float v = o[j][r] / fmaxf(lrow[r], 1e-30f);
      O[(size_t)(b * Ll + row) * Dd + col] = f2bf(v);
    }
}

// ---------------------------------------------------------------------------
extern "C" void kernel_launch(void* const* d_in, const int* in_sizes, int n_in,
                              void* d_out, int out_size, void* d_ws, size_t ws_size,
                              hipStream_t stream) {
  const float* x    = (const float*)d_in[0];
  const float* subj = (const float*)d_in[1];
  const unsigned char* mask = (const unsigned char*)d_in[2];
  const float* Wq  = (const float*)d_in[3];
  const float* Wk  = (const float*)d_in[4];
  const float* Wv  = (const float*)d_in[5];
  const float* Wo  = (const float*)d_in[6];
  const float* bo  = (const float*)d_in[7];
  const float* Wsq = (const float*)d_in[8];
  const float* bsq = (const float*)d_in[9];
  const float* Wsk = (const float*)d_in[10];
  const float* bsk = (const float*)d_in[11];
  float* out = (float*)d_out;

  char* ws = (char*)d_ws;
  size_t off = 0;
  auto walloc = [&](size_t bytes) -> void* {
    void* p = ws + off;
    off += (bytes + 255) & ~(size_t)255;
    return p;
  };
  const size_t nX = (size_t)Bb * Ll * Dd;     // 4,194,304
  const size_t nW = (size_t)Dd * Dd;          // 1,048,576

  bf16_t* xb  = (bf16_t*)walloc(nX * 2);
  bf16_t* wqb = (bf16_t*)walloc(nW * 2);
  bf16_t* wkb = (bf16_t*)walloc(nW * 2);
  bf16_t* wvb = (bf16_t*)walloc(nW * 2);
  bf16_t* wob = (bf16_t*)walloc(nW * 2);
  bf16_t* qh  = (bf16_t*)walloc(nX * 2);      // [B,H,L,64]
  bf16_t* kh  = (bf16_t*)walloc(nX * 2);
  bf16_t* vh  = (bf16_t*)walloc(nX * 2);
  bf16_t* ao  = (bf16_t*)walloc(nX * 2);      // [B,L,D]
  float*  qb  = (float*)walloc((size_t)Bb * Dd * 4);
  float*  kb  = (float*)walloc((size_t)Bb * Dd * 4);

  // 1) fp32 -> bf16 converts
  cvt_bf16_kernel<<<(int)(nX / 256), 256, 0, stream>>>(x, xb, (int)nX);
  cvt_bf16_kernel<<<(int)(nW / 256), 256, 0, stream>>>(Wq, wqb, (int)nW);
  cvt_bf16_kernel<<<(int)(nW / 256), 256, 0, stream>>>(Wk, wkb, (int)nW);
  cvt_bf16_kernel<<<(int)(nW / 256), 256, 0, stream>>>(Wv, wvb, (int)nW);
  cvt_bf16_kernel<<<(int)(nW / 256), 256, 0, stream>>>(Wo, wob, (int)nW);

  // 2) subject biases
  subject_bias_kernel<<<(Bb * Dd + 255) / 256, 256, 0, stream>>>(
      subj, Wsq, bsq, Wsk, bsk, qb, kb);

  // 3) Q/K/V projections (fused bias), head layout out
  const int M = Bb * Ll, N = Dd, K = Dd;
  const int gemmGrid = (M / 64) * (N / 64);   // 1024
  gemm_bf16_kernel<<<gemmGrid, 128, 0, stream>>>(xb, wqb, qb, qh, M, N, K, 0);
  gemm_bf16_kernel<<<gemmGrid, 128, 0, stream>>>(xb, wkb, kb, kh, M, N, K, 0);
  gemm_bf16_kernel<<<gemmGrid, 128, 0, stream>>>(xb, wvb, nullptr, vh, M, N, K, 0);

  // 4) flash attention: 32 (b,h) * 32 query blocks
  flash_attn_kernel<<<Bb * Hh * (Ll / 64), 128, 0, stream>>>(qh, kh, vh, mask, ao);

  // 5) output projection: out = ao @ Wo + bo  (fp32 out)
  gemm_bf16_kernel<<<gemmGrid, 128, 0, stream>>>(ao, wob, bo, out, M, N, K, 1);
}
